// MultiHeadGATLayer_1597727834396
// MI455X (gfx1250) — matrix operand order repaired and checked by reading, not compiled, not run on hardware
//
#include <hip/hip_runtime.h>
#include <hip/hip_bf16.h>
#include <math.h>

typedef __attribute__((ext_vector_type(16))) _Float16 v16h;
typedef __attribute__((ext_vector_type(8)))  float    v8f;

constexpr int INC = 128;   // input channels
constexpr int HC  = 128;   // H*C output channels
constexpr int H   = 8;
constexpr int C   = 16;
constexpr float NEG_SLOPE = 0.2f;

// ---------- monotone float <-> uint key for atomic segment-max ----------
__device__ __forceinline__ unsigned f2key(float f) {
    unsigned b = __float_as_uint(f);
    return (b & 0x80000000u) ? ~b : (b | 0x80000000u);
}
__device__ __forceinline__ float key2f(unsigned k) {
    unsigned b = (k & 0x80000000u) ? (k & 0x7FFFFFFFu) : ~k;
    return __uint_as_float(b);
}

// ---------- K0: init out = bias, amax keys = 0 (== -NaN sentinel), denom = 0 ----------
__global__ void k_init(float* __restrict__ out, const float* __restrict__ bias,
                       unsigned* __restrict__ amaxKey, float* __restrict__ denom, int n) {
    int idx = blockIdx.x * blockDim.x + threadIdx.x;
    if (idx < n * HC) out[idx] = bias[idx & (HC - 1)];
    if (idx < n * H) { amaxKey[idx] = 0u; denom[idx] = 0.0f; }
}

// ---------- K1: xt = x @ W^T via WMMA (f16 hi/lo split, f32 accumulate) ----------
// One 16(M)x16(N) tile per wave; 8 waves/block cover all 8 N-tiles of one M-tile.
__global__ __launch_bounds__(256) void k_gemm_wmma(const float* __restrict__ x,
                                                   const float* __restrict__ W,
                                                   float* __restrict__ xt, int n) {
    const int lane = threadIdx.x & 31;
    const int wave = threadIdx.x >> 5;      // 0..7 -> N-tile
    const int m0 = blockIdx.x * 16;
    const int n0 = wave * 16;

    const int r      = lane & 15;
    const int khalfA = (lane >> 4) << 3;    // 0 or 8   (A frag K sub-offset)
    const int koffB  = (lane >> 4) << 4;    // 0 or 16  (unused; B frag also uses khalfA)

    int row = m0 + r;
    if (row >= n) row = n - 1;              // clamp (n is a multiple of 16 anyway)
    const float* xr = x + (size_t)row * INC;
    const float* wr = W + (size_t)(n0 + r) * INC;   // B[k, col] = W[col, k]

    v8f acc = {};
#pragma unroll
    for (int kt = 0; kt < INC; kt += 32) {
        float af[16], bf[16];
#pragma unroll
        for (int j = 0; j < 8; ++j) {
            af[j]     = xr[kt + khalfA + j];        // frag elems 0..7  : K = kt+khalfA+j
            af[8 + j] = xr[kt + 16 + khalfA + j];   // frag elems 8..15 : K = kt+16+khalfA+j
        }
#pragma unroll
        for (int j = 0; j < 8; ++j) {
            bf[j]     = wr[kt + khalfA + j];        // frag elems 0..7  : K = kt+khalfA+j
            bf[8 + j] = wr[kt + 16 + khalfA + j];   // frag elems 8..15 : K = kt+16+khalfA+j
        }

        v16h ah, al, bh, bl;
#pragma unroll
        for (int j = 0; j < 16; ++j) {
            _Float16 h0 = (_Float16)af[j];
            ah[j] = h0;
            al[j] = (_Float16)(af[j] - (float)h0);
            _Float16 h1 = (_Float16)bf[j];
            bh[j] = h1;
            bl[j] = (_Float16)(bf[j] - (float)h1);
        }
        // acc += A_lo*B_hi + A_hi*B_lo + A_hi*B_hi  (~f32 accuracy)
        acc = __builtin_amdgcn_wmma_f32_16x16x32_f16(false, al, false, bh, (short)0, acc, false, false);
        acc = __builtin_amdgcn_wmma_f32_16x16x32_f16(false, ah, false, bl, (short)0, acc, false, false);
        acc = __builtin_amdgcn_wmma_f32_16x16x32_f16(false, ah, false, bh, (short)0, acc, false, false);
    }

    // D layout: VGPR i -> row m0+i (lanes 0-15) / m0+i+8 (lanes 16-31), col = lane&15
    const int rbase = m0 + ((lane >> 4) << 3);
#pragma unroll
    for (int i = 0; i < 8; ++i) {
        int ro = rbase + i;
        if (ro < n) xt[(size_t)ro * HC + n0 + r] = acc[i];
    }
}

// ---------- K2: per-node attention dots s[n,h] = <xt[n,h,:],att_s>, d likewise ----------
__global__ void k_node_sd(const float* __restrict__ xt, const float* __restrict__ att,
                          float* __restrict__ s, float* __restrict__ d, int n) {
    int idx = blockIdx.x * blockDim.x + threadIdx.x;
    if (idx >= n * H) return;
    int node = idx / H, h = idx - node * H;
    const float* v  = xt + (size_t)node * HC + h * C;
    const float* as = att + h * 2 * C;       // att[0,h,0:16]
    const float* ad = as + C;                // att[0,h,16:32]
    float ss = 0.f, dd = 0.f;
#pragma unroll
    for (int c = 0; c < C; ++c) { ss += v[c] * as[c]; dd += v[c] * ad[c]; }
    s[idx] = ss; d[idx] = dd;
}

// ---------- K3: edge logits + atomic segment-max ----------
__global__ void k_edge_alpha(const int* __restrict__ ei, const float* __restrict__ s,
                             const float* __restrict__ d, float* __restrict__ alpha,
                             unsigned* __restrict__ amaxKey, int E, int n) {
    int idx = blockIdx.x * blockDim.x + threadIdx.x;
    if (idx >= (E + n) * H) return;
    int e = idx / H, h = idx - e * H;
    int src, tgt;
    if (e < E) { src = ei[e]; tgt = ei[E + e]; }
    else       { src = tgt = e - E; }               // self-loop
    float a = s[src * H + h] + d[tgt * H + h];
    a = (a > 0.f) ? a : NEG_SLOPE * a;              // leaky relu
    alpha[idx] = a;
    atomicMax(&amaxKey[tgt * H + h], f2key(a));
}

// ---------- K4: exp(alpha - max) + atomic segment-sum of denominators ----------
__global__ void k_edge_exp(const int* __restrict__ ei, float* __restrict__ alpha,
                           const unsigned* __restrict__ amaxKey, float* __restrict__ denom,
                           int E, int n) {
    int idx = blockIdx.x * blockDim.x + threadIdx.x;
    if (idx >= (E + n) * H) return;
    int e = idx / H, h = idx - e * H;
    int tgt = (e < E) ? ei[E + e] : (e - E);
    float am = key2f(amaxKey[tgt * H + h]);
    float ex = __expf(alpha[idx] - am);
    alpha[idx] = ex;                                // overwrite logits with exp
    atomicAdd(&denom[tgt * H + h], ex);
}

// ---------- K5: weighted scatter of messages; 1 wave per edge, float4 per lane ----------
__global__ __launch_bounds__(256) void k_edge_scatter(const int* __restrict__ ei,
                                                      const float* __restrict__ xt,
                                                      const float* __restrict__ exv,
                                                      const float* __restrict__ denom,
                                                      float* __restrict__ out, int E, int n) {
    int gid = blockIdx.x * blockDim.x + threadIdx.x;
    int e = gid >> 5;
    int lane = gid & 31;
    if (e >= E + n) return;
    int src, tgt;
    if (e < E) { src = ei[e]; tgt = ei[E + e]; }
    else       { src = tgt = e - E; }
    int h = lane >> 2;                              // 4 feats/lane, 16 feats/head
    float w = exv[(size_t)e * H + h] / denom[tgt * H + h];
    const float4 v = *(const float4*)(xt + (size_t)src * HC + lane * 4);
    float* o = out + (size_t)tgt * HC + lane * 4;
    atomicAdd(o + 0, w * v.x);
    atomicAdd(o + 1, w * v.y);
    atomicAdd(o + 2, w * v.z);
    atomicAdd(o + 3, w * v.w);
}

extern "C" void kernel_launch(void* const* d_in, const int* in_sizes, int n_in,
                              void* d_out, int out_size, void* d_ws, size_t ws_size,
                              hipStream_t stream) {
    const float* x    = (const float*)d_in[0];
    const int*   ei   = (const int*)d_in[1];
    const float* W    = (const float*)d_in[2];
    const float* att  = (const float*)d_in[3];
    const float* bias = (const float*)d_in[4];
    float* out = (float*)d_out;

    const int n = in_sizes[0] / INC;   // 50000 nodes
    const int E = in_sizes[1] / 2;     // 400000 edges
    const int Etot = E + n;

    // workspace carve-up (all float-aligned): ~46.4 MB total
    char* ws = (char*)d_ws;
    float*    xt      = (float*)ws;     ws += (size_t)n * HC * sizeof(float);
    float*    sbuf    = (float*)ws;     ws += (size_t)n * H  * sizeof(float);
    float*    dbuf    = (float*)ws;     ws += (size_t)n * H  * sizeof(float);
    unsigned* amaxKey = (unsigned*)ws;  ws += (size_t)n * H  * sizeof(unsigned);
    float*    denom   = (float*)ws;     ws += (size_t)n * H  * sizeof(float);
    float*    alpha   = (float*)ws;     // (E+n)*H floats

    k_init<<<(n * HC + 255) / 256, 256, 0, stream>>>(out, bias, amaxKey, denom, n);
    k_gemm_wmma<<<(n + 15) / 16, 256, 0, stream>>>(x, W, xt, n);
    k_node_sd<<<(n * H + 255) / 256, 256, 0, stream>>>(xt, att, sbuf, dbuf, n);
    k_edge_alpha<<<(Etot * H + 255) / 256, 256, 0, stream>>>(ei, sbuf, dbuf, alpha, amaxKey, E, n);
    k_edge_exp<<<(Etot * H + 255) / 256, 256, 0, stream>>>(ei, alpha, amaxKey, denom, E, n);
    k_edge_scatter<<<(Etot * 32 + 255) / 256, 256, 0, stream>>>(ei, xt, alpha, denom, out, E, n);
}
